// H2GCN_738734375591
// MI455X (gfx1250) — matrix operand order repaired and checked by reading, not compiled, not run on hardware
//
#include <hip/hip_runtime.h>
#include <hip/hip_bf16.h>
#include <math.h>

// Problem constants (match reference)
#define NN    50000
#define F_IN  512
#define HID   64
#define CC    32
#define KCAT  192   // 3*HID concat width

typedef __attribute__((ext_vector_type(16))) __bf16 v16bf;
typedef __attribute__((ext_vector_type(8)))  float  v8f;

#define WMMA_BF16(a, b, c) \
    __builtin_amdgcn_wmma_f32_16x16x32_bf16(false, (a), false, (b), (short)0, (c), false, false)

// Build a 16x32 bf16 A-fragment for v_wmma_f32_16x16x32_bf16 from a
// row-major fp32 row segment. Contiguous k order, same as the B staging (wave32):
//   lane l (m=l%16, h=l/16) holds K = 16h+i (vec pos i = 0..15)
__device__ __forceinline__ v16bf load_a_frag(const float* __restrict__ base, int h) {
    v16bf a;
    const float* p0 = base + 16 * h;
    const float* p1 = base + 16 * h + 8;
#pragma unroll
    for (int i = 0; i < 8; ++i) a[i] = (__bf16)p0[i];
#pragma unroll
    for (int i = 0; i < 8; ++i) a[8 + i] = (__bf16)p1[i];
    return a;
}

// -------------------- GEMM1: R0 = relu(x @ W1 + b1) --------------------
// 256 threads = 8 waves; each wave computes a 16x64 tile (4 WMMA N-tiles).
// W1 (512x64) staged once per block into LDS, bf16, pre-swizzled into
// B-fragment order: lane l (n=l%16, h=l/16) holds B[ktile*32 + 16h + i][n].
__global__ __launch_bounds__(256) void gemm1_kernel(
    const float* __restrict__ x, const float* __restrict__ W1,
    const float* __restrict__ b1, float* __restrict__ Rcat) {
    __shared__ __align__(32) __bf16 w1s[16 * 4 * 32 * 16]; // [kt][nt][lane][i] = 64 KB
    const int tid = threadIdx.x;

    for (int f = tid; f < 16 * 4 * 32 * 16; f += 256) {
        int i     = f & 15;
        int lane  = (f >> 4) & 31;
        int ntile = (f >> 9) & 3;
        int ktile = f >> 11;
        int n = ntile * 16 + (lane & 15);
        int h = lane >> 4;
        int k = ktile * 32 + 16 * h + i;
        w1s[f] = (__bf16)W1[k * HID + n];
    }
    __syncthreads();

    const int wave = tid >> 5;
    const int lane = tid & 31;
    const int m    = lane & 15;
    const int h    = lane >> 4;
    const int row0 = blockIdx.x * 128 + wave * 16;

    int rowm = row0 + m;
    if (rowm >= NN) rowm = NN - 1;           // clamp: EXEC must stay all-ones for WMMA
    const float* xr = x + (size_t)rowm * F_IN;

    v8f zf = {0.f, 0.f, 0.f, 0.f, 0.f, 0.f, 0.f, 0.f};
    v8f acc[4] = {zf, zf, zf, zf};

    for (int kt = 0; kt < 16; ++kt) {
        v16bf a = load_a_frag(xr + kt * 32, h);
        // Batch all four B fragments first (one ds clause, one wait),
        // then issue 4 back-to-back WMMAs sharing the A operand.
        const __bf16* bp = &w1s[(kt * 128 + lane) * 16];
        v16bf b0 = *(const v16bf*)(bp + 0 * 512);
        v16bf b1 = *(const v16bf*)(bp + 1 * 512);
        v16bf b2 = *(const v16bf*)(bp + 2 * 512);
        v16bf b3 = *(const v16bf*)(bp + 3 * 512);
        acc[0] = WMMA_BF16(a, b0, acc[0]);
        acc[1] = WMMA_BF16(a, b1, acc[1]);
        acc[2] = WMMA_BF16(a, b2, acc[2]);
        acc[3] = WMMA_BF16(a, b3, acc[3]);
    }

    // D layout: lane holds D[r + 8h][m] in acc[nt][r]; col = nt*16 + m
#pragma unroll
    for (int nt = 0; nt < 4; ++nt) {
        int   colv = nt * 16 + m;
        float bias = b1[colv];
#pragma unroll
        for (int r = 0; r < 8; ++r) {
            int row = row0 + r + 8 * h;
            if (row < NN) {
                float v = acc[nt][r] + bias;
                Rcat[(size_t)row * KCAT + colv] = v > 0.f ? v : 0.f;
            }
        }
    }
}

// -------------------- SpMM: Rout[row] += val * Rin[col] --------------------
// 16 lanes per edge, float4 gather (Rin is L2-resident, 12.8 MB), f32 atomics.
// Edge lists are the only real HBM stream -> prefetch them ahead.
__global__ __launch_bounds__(256) void spmm_kernel(
    const int* __restrict__ erow, const int* __restrict__ ecol,
    const float* __restrict__ eval, const float* __restrict__ Rin, int inStride,
    float* __restrict__ Rout, int E) {
    int t   = blockIdx.x * blockDim.x + threadIdx.x;
    int e   = t >> 4;
    int l16 = t & 15;
    if (e >= E) return;
    if (l16 == 0 && e + 2048 < E) {          // global_prefetch_b8 of the edge stream
        __builtin_prefetch(erow + e + 2048, 0, 0);
        __builtin_prefetch(ecol + e + 2048, 0, 0);
        __builtin_prefetch(eval + e + 2048, 0, 0);
    }
    float v = eval[e];
    if (v == 0.f) return;                    // padded edges contribute nothing
    int c = ecol[e];
    int r = erow[e];
    const float4 g = *(const float4*)(Rin + (size_t)c * inStride + l16 * 4);
    float* o = Rout + (size_t)r * HID + l16 * 4;
    atomicAdd(o + 0, v * g.x);
    atomicAdd(o + 1, v * g.y);
    atomicAdd(o + 2, v * g.z);
    atomicAdd(o + 3, v * g.w);
}

// -------------------- max-combine into concat buffer --------------------
__global__ __launch_bounds__(256) void maxcomb_kernel(
    const float* __restrict__ R1, const float* __restrict__ R2,
    float* __restrict__ dst /* Rcat + col offset */, int n) {
    int i = blockIdx.x * blockDim.x + threadIdx.x;
    if (i >= n) return;
    int node = i >> 6;
    int f    = i & 63;
    dst[(size_t)node * KCAT + f] = fmaxf(R1[i], R2[i]);
}

__global__ __launch_bounds__(256) void zero_kernel(float4* __restrict__ p, int n4) {
    int i = blockIdx.x * blockDim.x + threadIdx.x;
    if (i < n4) p[i] = make_float4(0.f, 0.f, 0.f, 0.f);
}

// -------------------- GEMM2 + log_softmax --------------------
__global__ __launch_bounds__(256) void gemm2_kernel(
    const float* __restrict__ Rcat, const float* __restrict__ W2,
    const float* __restrict__ b2, float* __restrict__ out) {
    __shared__ __align__(32) __bf16 w2s[6 * 2 * 32 * 16]; // 12 KB
    __shared__ float lg[8][16][CC];                       // 16 KB logits scratch
    const int tid = threadIdx.x;

    for (int f = tid; f < 6 * 2 * 32 * 16; f += 256) {
        int i     = f & 15;
        int lane  = (f >> 4) & 31;
        int ntile = (f >> 9) & 1;
        int ktile = f >> 10;
        int n = ntile * 16 + (lane & 15);
        int h = lane >> 4;
        int k = ktile * 32 + 16 * h + i;
        w2s[f] = (__bf16)W2[k * CC + n];
    }
    __syncthreads();

    const int wave = tid >> 5;
    const int lane = tid & 31;
    const int m    = lane & 15;
    const int h    = lane >> 4;
    const int row0 = blockIdx.x * 128 + wave * 16;

    int rowm = row0 + m;
    if (rowm >= NN) rowm = NN - 1;
    const float* rr = Rcat + (size_t)rowm * KCAT;

    v8f zf = {0.f, 0.f, 0.f, 0.f, 0.f, 0.f, 0.f, 0.f};
    v8f acc[2] = {zf, zf};

#pragma unroll
    for (int kt = 0; kt < 6; ++kt) {
        v16bf a = load_a_frag(rr + kt * 32, h);
        const __bf16* bp = &w2s[(kt * 64 + lane) * 16];
        v16bf b0 = *(const v16bf*)(bp + 0 * 512);
        v16bf b1 = *(const v16bf*)(bp + 1 * 512);
        acc[0] = WMMA_BF16(a, b0, acc[0]);
        acc[1] = WMMA_BF16(a, b1, acc[1]);
    }

#pragma unroll
    for (int nt = 0; nt < 2; ++nt) {
        int   colv = nt * 16 + m;
        float bias = b2[colv];
#pragma unroll
        for (int r = 0; r < 8; ++r)
            lg[wave][r + 8 * h][colv] = acc[nt][r] + bias;
    }
    __syncthreads();

    if (lane < 16) {
        int row = row0 + lane;
        if (row < NN) {
            const float* v = lg[wave][lane];
            float mx = v[0];
#pragma unroll
            for (int c = 1; c < CC; ++c) mx = fmaxf(mx, v[c]);
            float s = 0.f;
#pragma unroll
            for (int c = 0; c < CC; ++c) s += expf(v[c] - mx);
            float lz = mx + logf(s);
            float* o = out + (size_t)row * CC;
#pragma unroll
            for (int c = 0; c < CC; ++c) o[c] = v[c] - lz;
        }
    }
}

extern "C" void kernel_launch(void* const* d_in, const int* in_sizes, int n_in,
                              void* d_out, int out_size, void* d_ws, size_t ws_size,
                              hipStream_t stream) {
    const float* x   = (const float*)d_in[0];
    const int*   e1r = (const int*)  d_in[1];
    const int*   e1c = (const int*)  d_in[2];
    const float* e1v = (const float*)d_in[3];
    const int*   e2r = (const int*)  d_in[4];
    const int*   e2c = (const int*)  d_in[5];
    const float* e2v = (const float*)d_in[6];
    const float* W1  = (const float*)d_in[7];
    const float* b1  = (const float*)d_in[8];
    const float* W2  = (const float*)d_in[9];
    const float* b2  = (const float*)d_in[10];
    float* out = (float*)d_out;

    const int E1 = in_sizes[1];
    const int E2 = in_sizes[4];

    char*  ws   = (char*)d_ws;
    float* Rcat = (float*)ws;                               // NN x 192
    float* R1   = (float*)(ws + (size_t)NN * KCAT * 4);     // NN x 64
    float* R2   = R1 + (size_t)NN * HID;                    // NN x 64 (contiguous after R1)

    const int NB = (NN + 127) / 128;                        // 391 row-tile blocks

    // Stage 1: R0 = relu(x @ W1 + b1) -> Rcat[:, 0:64]
    gemm1_kernel<<<NB, 256, 0, stream>>>(x, W1, b1, Rcat);

    // Stages 2-3: two rounds of max(SpMM(A1,R), SpMM(A2,R))
    for (int round = 0; round < 2; ++round) {
        const int nz4 = (2 * NN * HID) / 4;                 // zero R1 and R2 together
        zero_kernel<<<(nz4 + 255) / 256, 256, 0, stream>>>((float4*)R1, nz4);

        const float* Rin = Rcat + round * HID;              // previous round's columns
        spmm_kernel<<<((size_t)E1 * 16 + 255) / 256, 256, 0, stream>>>(
            e1r, e1c, e1v, Rin, KCAT, R1, E1);
        spmm_kernel<<<((size_t)E2 * 16 + 255) / 256, 256, 0, stream>>>(
            e2r, e2c, e2v, Rin, KCAT, R2, E2);

        maxcomb_kernel<<<(NN * HID + 255) / 256, 256, 0, stream>>>(
            R1, R2, Rcat + (round + 1) * HID, NN * HID);
    }

    // Stage 4: log_softmax(Rcat @ W2 + b2)
    gemm2_kernel<<<NB, 256, 0, stream>>>(Rcat, W2, b2, out);
}
